// Attention_31602369364414
// MI455X (gfx1250) — matrix operand order repaired and checked by reading, not compiled, not run on hardware
//
#include <hip/hip_runtime.h>
#include <hip/hip_bf16.h>

// ---------------------------------------------------------------------------
// Two-stage attention for MI455X (gfx1250, wave32, WMMA 16x16x32 f16->f32)
//   B=8, N=1024, DIM=512, H=8, D=64
// ---------------------------------------------------------------------------

typedef __attribute__((ext_vector_type(16))) _Float16 v16h;
typedef __attribute__((ext_vector_type(8)))  float    v8f;

#define ATT_B    8
#define ATT_N    1024
#define ATT_DIM  512
#define ATT_H    8
#define ATT_D    64
#define QKV_STR  (3 * ATT_DIM)          // 1536
#define ATT_SCALE 0.044194173824159216f // DIM^-0.5

// ---- WMMA wrapper ----------------------------------------------------------
__device__ __forceinline__ v8f wmma_f16(v16h a, v16h b, v8f c) {
    // D = A(16x32 f16) * B(32x16 f16) + C(16x16 f32)
    return __builtin_amdgcn_wmma_f32_16x16x32_f16(
        /*neg_a=*/false, a, /*neg_b=*/false, b,
        /*c_mod=*/(short)0, c, /*reuse_a=*/false, /*reuse_b=*/false);
}

// ---- Fragment loaders (contiguous k = 16*half + e order, same for A and B), f32 source -> f16 fragment -----
// A 16x32 (MxK): lane 0-15 -> M=lane, element e -> K=kh*16+e
__device__ __forceinline__ v16h load_a_frag_f32(const float* __restrict__ src, int ld) {
    const int lane = threadIdx.x & 31;
    const int m    = lane & 15;
    const int kh   = lane >> 4;          // 0 or 1
    const float* row = src + (size_t)m * ld;
    v16h a;
#pragma unroll
    for (int e = 0; e < 16; ++e) a[e] = (_Float16)row[kh * 16 + e];
    return a;
}

// A fragment from an f16 LDS tile (row-major 16 x 32)
__device__ __forceinline__ v16h load_a_frag_lds(const _Float16* __restrict__ src, int ld) {
    const int lane = threadIdx.x & 31;
    const int m    = lane & 15;
    const int kh   = lane >> 4;
    const _Float16* row = src + m * ld;
    v16h a;
#pragma unroll
    for (int e = 0; e < 16; ++e) a[e] = row[kh * 16 + e];
    return a;
}

// B 32x16 (KxN), row-major B[k][n] source: lane = N col, element e -> K=kg*16+e
__device__ __forceinline__ v16h load_b_frag_f32(const float* __restrict__ src, int ld) {
    const int lane = threadIdx.x & 31;
    const int n    = lane & 15;
    const int kg   = lane >> 4;          // 0 or 1
    v16h b;
#pragma unroll
    for (int e = 0; e < 16; ++e)
        b[e] = (_Float16)src[(size_t)(kg * 16 + e) * ld + n];
    return b;
}

// B fragment where B[k][n] = M[n][k]  (i.e. K^T: keys are rows, d is contiguous)
__device__ __forceinline__ v16h load_bT_frag_f32(const float* __restrict__ src, int ld) {
    const int lane = threadIdx.x & 31;
    const int n    = lane & 15;
    const int kg   = lane >> 4;
    const float* row = src + (size_t)n * ld;
    v16h b;
#pragma unroll
    for (int e = 0; e < 16; ++e)
        b[e] = (_Float16)row[kg * 16 + e];
    return b;
}

// ---------------------------------------------------------------------------
// GEMM + bias:  C[M x Nt] = A[M x K] * W[K x Nt] + bias[Nt]   (f32 in/out)
// One wave computes a 32x32 block (2x2 WMMA tiles).
// ---------------------------------------------------------------------------
__global__ void __launch_bounds__(256)
gemm_bias_kernel(const float* __restrict__ A, const float* __restrict__ W,
                 const float* __restrict__ bias, float* __restrict__ C,
                 int M, int K, int Nt) {
    const int wave    = (int)((blockIdx.x * blockDim.x + threadIdx.x) >> 5);
    const int tiles_n = Nt >> 5;
    const int m0 = (wave / tiles_n) * 32;
    const int n0 = (wave % tiles_n) * 32;
    const int lane = threadIdx.x & 31;

    v8f acc00 = {}, acc01 = {}, acc10 = {}, acc11 = {};

    for (int k = 0; k < K; k += 32) {
        // Prefetch next K-step: 32 A rows and 32 W rows (one row per lane)
        if (k + 32 < K) {
            __builtin_prefetch(A + (size_t)(m0 + (lane & 15)) * K + k + 32 + (lane >> 4) * 16);
            __builtin_prefetch(A + (size_t)(m0 + 16 + (lane & 15)) * K + k + 32 + (lane >> 4) * 16);
            __builtin_prefetch(W + (size_t)(k + 32 + lane) * Nt + n0);
        }
        v16h a0 = load_a_frag_f32(A + (size_t)m0 * K + k, K);
        v16h a1 = load_a_frag_f32(A + (size_t)(m0 + 16) * K + k, K);
        v16h b0 = load_b_frag_f32(W + (size_t)k * Nt + n0, Nt);
        v16h b1 = load_b_frag_f32(W + (size_t)k * Nt + n0 + 16, Nt);
        acc00 = wmma_f16(a0, b0, acc00);
        acc01 = wmma_f16(a0, b1, acc01);
        acc10 = wmma_f16(a1, b0, acc10);
        acc11 = wmma_f16(a1, b1, acc11);
    }

    const int n  = lane & 15;
    const int hi = lane >> 4;
    const float bv0 = bias[n0 + n];
    const float bv1 = bias[n0 + 16 + n];
#pragma unroll
    for (int r = 0; r < 8; ++r) {
        const int row0 = m0 + r + 8 * hi;
        const int row1 = row0 + 16;
        C[(size_t)row0 * Nt + n0 + n]      = acc00[r] + bv0;
        C[(size_t)row0 * Nt + n0 + 16 + n] = acc01[r] + bv1;
        C[(size_t)row1 * Nt + n0 + n]      = acc10[r] + bv0;
        C[(size_t)row1 * Nt + n0 + 16 + n] = acc11[r] + bv1;
    }
}

// ---------------------------------------------------------------------------
// Stage 1: out1 = sigmoid((Q K^T) * mask) @ V      (no scale, no softmax)
// One wave per (b, h, 16-query tile). Processes keys 32 at a time.
// ---------------------------------------------------------------------------
__global__ void __launch_bounds__(256)
attn_sigmoid_kernel(const float* __restrict__ qkv, const float* __restrict__ mask,
                    float* __restrict__ out1) {
    __shared__ _Float16 pbuf[8 * 16 * 32];    // per-wave 16x32 f16 P-tile
    _Float16* pb = pbuf + (threadIdx.x >> 5) * (16 * 32);

    const int wid    = (int)((blockIdx.x * blockDim.x + threadIdx.x) >> 5);
    const int q_tile = wid % (ATT_N / 16);
    const int h      = (wid / (ATT_N / 16)) % ATT_H;
    const int b      = wid / ((ATT_N / 16) * ATT_H);
    const int i0     = q_tile * 16;

    const float* base = qkv + (size_t)b * ATT_N * QKV_STR;
    const float* Qp   = base + (size_t)i0 * QKV_STR + h * ATT_D;          // Q
    const float* Kb   = base + ATT_DIM + h * ATT_D;                       // K
    const float* Vb   = base + 2 * ATT_DIM + h * ATT_D;                   // V

    // Preload Q fragments for d = 0..31 and 32..63
    v16h qa0 = load_a_frag_f32(Qp, QKV_STR);
    v16h qa1 = load_a_frag_f32(Qp + 32, QKV_STR);

    v8f o[4] = {};                            // 16 x 64 output accumulator

    const int lane = threadIdx.x & 31;
    const int n    = lane & 15;
    const int hi   = lane >> 4;

    for (int j0 = 0; j0 < ATT_N; j0 += 32) {
        // Prefetch next key-block's K and V rows (one row per lane)
        if (j0 + 32 < ATT_N) {
            __builtin_prefetch(Kb + (size_t)(j0 + 32 + lane) * QKV_STR);
            __builtin_prefetch(Vb + (size_t)(j0 + 32 + lane) * QKV_STR);
        }
        // S = Q K^T over this 32-key block (two 16x16 tiles)
        const float* Kp = Kb + (size_t)j0 * QKV_STR;
        v8f s0 = {}, s1 = {};
        v16h kb00 = load_bT_frag_f32(Kp, QKV_STR);
        v16h kb10 = load_bT_frag_f32(Kp + (size_t)16 * QKV_STR, QKV_STR);
        s0 = wmma_f16(qa0, kb00, s0);
        s1 = wmma_f16(qa0, kb10, s1);
        v16h kb01 = load_bT_frag_f32(Kp + 32, QKV_STR);
        v16h kb11 = load_bT_frag_f32(Kp + (size_t)16 * QKV_STR + 32, QKV_STR);
        s0 = wmma_f16(qa1, kb01, s0);
        s1 = wmma_f16(qa1, kb11, s1);

        // p = sigmoid(s * mask);  stash as f16 tile in LDS (C-layout -> row major)
#pragma unroll
        for (int r = 0; r < 8; ++r) {
            const int mrow = r + 8 * hi;
            const int gi   = i0 + mrow;
            const float mv0 = mask[(size_t)gi * ATT_N + j0 + n];
            const float mv1 = mask[(size_t)gi * ATT_N + j0 + 16 + n];
            const float p0 = 1.0f / (1.0f + __expf(-s0[r] * mv0));
            const float p1 = 1.0f / (1.0f + __expf(-s1[r] * mv1));
            pb[mrow * 32 + n]      = (_Float16)p0;
            pb[mrow * 32 + 16 + n] = (_Float16)p1;
        }
        asm volatile("s_wait_dscnt 0" ::: "memory");   // LDS RAW fence (per-wave)

        // O += P (16x32) @ V (32x64)
        v16h pa = load_a_frag_lds(pb, 32);
        const float* Vp = Vb + (size_t)j0 * QKV_STR;
#pragma unroll
        for (int t = 0; t < 4; ++t) {
            v16h vb = load_b_frag_f32(Vp + t * 16, QKV_STR);
            o[t] = wmma_f16(pa, vb, o[t]);
        }
    }

    // out1[b, i0+row, h*64 + t*16 + n]
    float* op = out1 + (size_t)b * ATT_N * ATT_DIM + (size_t)i0 * ATT_DIM + h * ATT_D;
#pragma unroll
    for (int t = 0; t < 4; ++t)
#pragma unroll
        for (int r = 0; r < 8; ++r)
            op[(size_t)(r + 8 * hi) * ATT_DIM + t * 16 + n] = o[t][r];
}

// ---------------------------------------------------------------------------
// Stage 2: out2 = softmax((Q K^T) * SCALE) @ V     (flash-style online softmax)
// ---------------------------------------------------------------------------
__global__ void __launch_bounds__(256)
attn_softmax_kernel(const float* __restrict__ qkv, float* __restrict__ out2) {
    __shared__ _Float16 pbuf[8 * 16 * 32];
    _Float16* pb = pbuf + (threadIdx.x >> 5) * (16 * 32);

    const int wid    = (int)((blockIdx.x * blockDim.x + threadIdx.x) >> 5);
    const int q_tile = wid % (ATT_N / 16);
    const int h      = (wid / (ATT_N / 16)) % ATT_H;
    const int b      = wid / ((ATT_N / 16) * ATT_H);
    const int i0     = q_tile * 16;

    const float* base = qkv + (size_t)b * ATT_N * QKV_STR;
    const float* Qp   = base + (size_t)i0 * QKV_STR + h * ATT_D;
    const float* Kb   = base + ATT_DIM + h * ATT_D;
    const float* Vb   = base + 2 * ATT_DIM + h * ATT_D;

    v16h qa0 = load_a_frag_f32(Qp, QKV_STR);
    v16h qa1 = load_a_frag_f32(Qp + 32, QKV_STR);

    v8f   o[4] = {};
    float mrow[8], lrow[8];
#pragma unroll
    for (int r = 0; r < 8; ++r) { mrow[r] = -1.0e30f; lrow[r] = 0.0f; }

    const int lane = threadIdx.x & 31;

    for (int j0 = 0; j0 < ATT_N; j0 += 32) {
        if (j0 + 32 < ATT_N) {
            __builtin_prefetch(Kb + (size_t)(j0 + 32 + lane) * QKV_STR);
            __builtin_prefetch(Vb + (size_t)(j0 + 32 + lane) * QKV_STR);
        }
        const float* Kp = Kb + (size_t)j0 * QKV_STR;
        v8f s0 = {}, s1 = {};
        v16h kb00 = load_bT_frag_f32(Kp, QKV_STR);
        v16h kb10 = load_bT_frag_f32(Kp + (size_t)16 * QKV_STR, QKV_STR);
        s0 = wmma_f16(qa0, kb00, s0);
        s1 = wmma_f16(qa0, kb10, s1);
        v16h kb01 = load_bT_frag_f32(Kp + 32, QKV_STR);
        v16h kb11 = load_bT_frag_f32(Kp + (size_t)16 * QKV_STR + 32, QKV_STR);
        s0 = wmma_f16(qa1, kb01, s0);
        s1 = wmma_f16(qa1, kb11, s1);

        float alpha[8];
#pragma unroll
        for (int r = 0; r < 8; ++r) {
            float sv0 = s0[r] * ATT_SCALE;
            float sv1 = s1[r] * ATT_SCALE;
            // row max across the 16-lane N-group (both 16-col tiles)
            float tmax = fmaxf(sv0, sv1);
#pragma unroll
            for (int off = 1; off < 16; off <<= 1)
                tmax = fmaxf(tmax, __shfl_xor(tmax, off, 16));
            const float mnew = fmaxf(mrow[r], tmax);
            alpha[r] = __expf(mrow[r] - mnew);
            const float p0 = __expf(sv0 - mnew);
            const float p1 = __expf(sv1 - mnew);
            float psum = p0 + p1;
#pragma unroll
            for (int off = 1; off < 16; off <<= 1)
                psum += __shfl_xor(psum, off, 16);
            lrow[r] = lrow[r] * alpha[r] + psum;
            mrow[r] = mnew;
            const int mr = r + 8 * (lane >> 4);
            pb[mr * 32 + (lane & 15)]      = (_Float16)p0;
            pb[mr * 32 + 16 + (lane & 15)] = (_Float16)p1;
        }
        // rescale running output rows (row r of O matches row r of S)
#pragma unroll
        for (int t = 0; t < 4; ++t)
#pragma unroll
            for (int r = 0; r < 8; ++r)
                o[t][r] *= alpha[r];

        asm volatile("s_wait_dscnt 0" ::: "memory");

        v16h pa = load_a_frag_lds(pb, 32);
        const float* Vp = Vb + (size_t)j0 * QKV_STR;
#pragma unroll
        for (int t = 0; t < 4; ++t) {
            v16h vb = load_b_frag_f32(Vp + t * 16, QKV_STR);
            o[t] = wmma_f16(pa, vb, o[t]);
        }
    }

    float* op = out2 + (size_t)b * ATT_N * ATT_DIM + (size_t)i0 * ATT_DIM + h * ATT_D;
#pragma unroll
    for (int t = 0; t < 4; ++t)
#pragma unroll
        for (int r = 0; r < 8; ++r)
            op[(size_t)(r + 8 * (lane >> 4)) * ATT_DIM + t * 16 + (lane & 15)] =
                o[t][r] * (1.0f / lrow[r]);
}

// ---------------------------------------------------------------------------
extern "C" void kernel_launch(void* const* d_in, const int* in_sizes, int n_in,
                              void* d_out, int out_size, void* d_ws, size_t ws_size,
                              hipStream_t stream) {
    const float* x     = (const float*)d_in[0];
    const float* mask  = (const float*)d_in[1];
    const float* Wqkv1 = (const float*)d_in[2];
    const float* bqkv1 = (const float*)d_in[3];
    const float* Wqkv2 = (const float*)d_in[4];
    const float* bqkv2 = (const float*)d_in[5];
    const float* Wnn1  = (const float*)d_in[6];
    const float* bnn1  = (const float*)d_in[7];
    float* out = (float*)d_out;

    const int M = ATT_B * ATT_N;          // 8192 rows

    float* qkv1 = (float*)d_ws;                               // [8192 x 1536]
    float* o1   = qkv1 + (size_t)M * QKV_STR;                 // [8192 x 512]
    float* qkv2 = o1 + (size_t)M * ATT_DIM;                   // [8192 x 1536]
    float* o2   = qkv1;                                       // reuse qkv1 (dead)

    const dim3 blk(256);

    // qkv1 = x @ Wqkv1 + bqkv1        waves = (8192/32)*(1536/32) = 12288
    gemm_bias_kernel<<<12288 / 8, blk, 0, stream>>>(x, Wqkv1, bqkv1, qkv1,
                                                    M, ATT_DIM, QKV_STR);
    // stage-1 sigmoid attention       waves = 8*8*64 = 4096
    attn_sigmoid_kernel<<<4096 / 8, blk, 0, stream>>>(qkv1, mask, o1);
    // qkv2 = out1 @ Wqkv2 + bqkv2
    gemm_bias_kernel<<<12288 / 8, blk, 0, stream>>>(o1, Wqkv2, bqkv2, qkv2,
                                                    M, ATT_DIM, QKV_STR);
    // stage-2 softmax attention
    attn_softmax_kernel<<<4096 / 8, blk, 0, stream>>>(qkv2, o2);
    // out = out2 @ Wnn1 + bnn1        waves = (8192/32)*(512/32) = 4096
    gemm_bias_kernel<<<4096 / 8, blk, 0, stream>>>(o2, Wnn1, bnn1, out,
                                                   M, ATT_DIM, ATT_DIM);
}
